// EdgeEncoder_1881195676157
// MI455X (gfx1250) — matrix operand order repaired and checked by reading, not compiled, not run on hardware
//
#include <hip/hip_runtime.h>
#include <hip/hip_bf16.h>

typedef __bf16 bf16_t;
typedef __attribute__((ext_vector_type(16))) __bf16 v16bf;
typedef __attribute__((ext_vector_type(8)))  __bf16 v8bf;
typedef __attribute__((ext_vector_type(4)))  __bf16 v4bf;
typedef __attribute__((ext_vector_type(8)))  float  v8f;

#define IN_DIM   128
#define FEAT_DIM 256   // 2*IN_DIM (K of GEMM1)
#define HID_DIM  256
#define LAT_DIM  64
#define ET       128   // edges (rows) per block
#define FSTR     264   // padded LDS row stride in bf16 (breaks bank conflicts)

// float -> bf16, round-to-nearest-even
__device__ __forceinline__ bf16_t f2bf(float f) {
    union { float f; unsigned u; } a; a.f = f;
    unsigned r = a.u + 0x7FFFu + ((a.u >> 16) & 1u);
    unsigned short h = (unsigned short)(r >> 16);
    return __builtin_bit_cast(bf16_t, h);
}

// 32 contiguous bytes -> one bf16 WMMA fragment (16 elems/lane)
__device__ __forceinline__ v16bf ldfrag32(const bf16_t* p) {
    v8bf lo = *(const v8bf*)p;
    v8bf hi = *(const v8bf*)(p + 8);
    return __builtin_shufflevector(lo, hi, 0,1,2,3,4,5,6,7,8,9,10,11,12,13,14,15);
}

// A-matrix 16x32 bf16 fragment in this code's contiguous k order (not ISA table order):
// lanes 0-15: row m=lane, elements 0-15 = K[16*hlf .. +15] (contiguous, matches B)
__device__ __forceinline__ v16bf lda(const bf16_t* rowPtr, int hlf) {
    v8bf lo = *(const v8bf*)(rowPtr + 16 * hlf);
    v8bf hi = *(const v8bf*)(rowPtr + 16 * hlf + 8);
    return __builtin_shufflevector(lo, hi, 0,1,2,3,4,5,6,7,8,9,10,11,12,13,14,15);
}

#define WMMA_BF16(a, b, c) \
    __builtin_amdgcn_wmma_f32_16x16x32_bf16(false, (a), false, (b), (short)0, (c), false, false)

// ---------------------------------------------------------------------------
// Kernel 0: W1[256k][256n] f32 -> W1t[256n][256k] bf16 ; W2[256k][64n] -> W2t[64n][256k]
// ---------------------------------------------------------------------------
__global__ void prep_weights(const float* __restrict__ W1, const float* __restrict__ W2,
                             bf16_t* __restrict__ W1t, bf16_t* __restrict__ W2t) {
    int i = blockIdx.x * 256 + threadIdx.x;
    if (i < FEAT_DIM * HID_DIM) {
        int n = i >> 8, k = i & 255;                 // W1t[n][k]
        W1t[n * FEAT_DIM + k] = f2bf(W1[k * HID_DIM + n]);
    }
    if (i < LAT_DIM * HID_DIM) {
        int n = i >> 8, k = i & 255;                 // W2t[n][k]
        W2t[n * HID_DIM + k] = f2bf(W2[k * LAT_DIM + n]);
    }
}

// ---------------------------------------------------------------------------
// Kernel 1: fused gather + GEMM1(relu) + GEMM2
// ---------------------------------------------------------------------------
__global__ void __launch_bounds__(256)
edge_encoder_kernel(const float* __restrict__ x,
                    const int* __restrict__ ei,          // [2, E]
                    const bf16_t* __restrict__ W1t,      // [256n][256k]
                    const float* __restrict__ b1,
                    const bf16_t* __restrict__ W2t,      // [64n][256k]
                    const float* __restrict__ b2,
                    float* __restrict__ out,             // [E, 64]
                    int nEdges) {
    __shared__ __align__(16) bf16_t feat[ET * FSTR];
    __shared__ __align__(16) bf16_t hid [ET * FSTR];

    const int tid    = threadIdx.x;
    const int wave   = tid >> 5;
    const int lane   = tid & 31;
    const int lane15 = lane & 15;
    const int hlf    = lane >> 4;
    const int eBase  = blockIdx.x * ET;

    // ---- Stage 1: gather edge features -> LDS (bf16) --------------------
    for (int i = tid; i < ET * 64; i += 256) {
        int el = i >> 6;            // local edge row
        int c4 = i & 63;            // which 4-float chunk of the 256 feats
        int e  = eBase + el;
        float4 v = make_float4(0.f, 0.f, 0.f, 0.f);
        if (e < nEdges) {
            int node = (c4 < 32) ? ei[e] : ei[nEdges + e];
            v = *(const float4*)(x + (size_t)node * IN_DIM + (size_t)(c4 & 31) * 4);
        }
        v4bf pk;
        pk[0] = f2bf(v.x); pk[1] = f2bf(v.y); pk[2] = f2bf(v.z); pk[3] = f2bf(v.w);
        *(v4bf*)(feat + el * FSTR + c4 * 4) = pk;   // one ds_store_b64
    }
    __syncthreads();

    // ---- Stage 2: hidden = relu(feat @ W1 + b1) -------------------------
    // wave owns N-tiles {2w, 2w+1}; K-outer, B reused across 8 M-tiles.
    // Software-pipelined: B prefetched one k-step ahead, A one m-step ahead.
    {
        const int nt0 = wave * 2;
        v8f c0[8], c1[8];
        #pragma unroll
        for (int m = 0; m < 8; ++m) { c0[m] = (v8f)(0.f); c1[m] = (v8f)(0.f); }

        const float bias0 = b1[nt0 * 16 + lane15];
        const float bias1 = b1[nt0 * 16 + 16 + lane15];
        const bf16_t* w0 = W1t + (size_t)(nt0 * 16 + lane15) * FEAT_DIM + 16 * hlf;
        const bf16_t* w1 = w0 + (size_t)16 * FEAT_DIM;
        const bf16_t* arow = feat + lane15 * FSTR;

        v16bf bf0 = ldfrag32(w0);
        v16bf bf1 = ldfrag32(w1);
        #pragma unroll
        for (int k = 0; k < 8; ++k) {
            v16bf nb0 = bf0, nb1 = bf1;
            if (k < 7) {                        // prefetch next k's B frags
                nb0 = ldfrag32(w0 + (k + 1) * 32);
                nb1 = ldfrag32(w1 + (k + 1) * 32);
            }
            v16bf a = lda(arow + k * 32, hlf);  // m = 0
            #pragma unroll
            for (int m = 0; m < 8; ++m) {
                v16bf an = a;
                if (m < 7)                      // prefetch next m's A frag
                    an = lda(arow + (m + 1) * 16 * FSTR + k * 32, hlf);
                c0[m] = WMMA_BF16(a, bf0, c0[m]);
                c1[m] = WMMA_BF16(a, bf1, c1[m]);
                a = an;
            }
            bf0 = nb0; bf1 = nb1;
        }
        // epilogue: bias + relu -> bf16 hidden in LDS (C/D layout: m = r + 8*hlf)
        #pragma unroll
        for (int m = 0; m < 8; ++m) {
            #pragma unroll
            for (int r = 0; r < 8; ++r) {
                int row = m * 16 + hlf * 8 + r;
                float u0 = c0[m][r] + bias0; u0 = u0 > 0.f ? u0 : 0.f;
                float u1 = c1[m][r] + bias1; u1 = u1 > 0.f ? u1 : 0.f;
                hid[row * FSTR + nt0 * 16 + lane15]      = f2bf(u0);
                hid[row * FSTR + nt0 * 16 + 16 + lane15] = f2bf(u1);
            }
        }
    }
    __syncthreads();

    // ---- Stage 3: out = hidden @ W2 + b2 --------------------------------
    // wave owns N-tile (w&3) and M-tiles [(w>>2)*4 .. +3]
    {
        const int nt2 = wave & 3;
        const int mB  = (wave >> 2) * 4;
        v8f c2[4];
        #pragma unroll
        for (int j = 0; j < 4; ++j) c2[j] = (v8f)(0.f);

        const float bias2 = b2[nt2 * 16 + lane15];
        const bf16_t* w2 = W2t + (size_t)(nt2 * 16 + lane15) * HID_DIM + 16 * hlf;
        const bf16_t* hrow = hid + (mB * 16 + lane15) * FSTR;

        v16bf b = ldfrag32(w2);
        #pragma unroll
        for (int k = 0; k < 8; ++k) {
            v16bf nb = b;
            if (k < 7) nb = ldfrag32(w2 + (k + 1) * 32);
            v16bf a = lda(hrow + k * 32, hlf);  // j = 0
            #pragma unroll
            for (int j = 0; j < 4; ++j) {
                v16bf an = a;
                if (j < 3) an = lda(hrow + (j + 1) * 16 * FSTR + k * 32, hlf);
                c2[j] = WMMA_BF16(a, b, c2[j]);
                a = an;
            }
            b = nb;
        }
        // epilogue: +b2, non-temporal f32 stores (output is never re-read;
        // keep x/W1t resident in the 192MB L2 instead)
        #pragma unroll
        for (int j = 0; j < 4; ++j) {
            #pragma unroll
            for (int r = 0; r < 8; ++r) {
                int row = (mB + j) * 16 + hlf * 8 + r;
                int e = eBase + row;
                if (e < nEdges) {
                    float val = c2[j][r] + bias2;
                    __builtin_nontemporal_store(val, out + (size_t)e * LAT_DIM + nt2 * 16 + lane15);
                }
            }
        }
    }
}

extern "C" void kernel_launch(void* const* d_in, const int* in_sizes, int n_in,
                              void* d_out, int out_size, void* d_ws, size_t ws_size,
                              hipStream_t stream) {
    const float* x  = (const float*)d_in[0];
    const int*   ei = (const int*)d_in[1];
    const float* W1 = (const float*)d_in[2];
    const float* b1 = (const float*)d_in[3];
    const float* W2 = (const float*)d_in[4];
    const float* b2 = (const float*)d_in[5];
    float* out = (float*)d_out;

    const int nEdges = in_sizes[1] / 2;

    bf16_t* W1t = (bf16_t*)d_ws;                                   // 128 KB
    bf16_t* W2t = (bf16_t*)((char*)d_ws + FEAT_DIM * HID_DIM * 2); // +32 KB

    prep_weights<<<(FEAT_DIM * HID_DIM + 255) / 256, 256, 0, stream>>>(W1, W2, W1t, W2t);

    int grid = (nEdges + ET - 1) / ET;
    edge_encoder_kernel<<<grid, 256, 0, stream>>>(x, ei, W1t, b1, W2t, b2, out, nEdges);
}
